// Predictor_7911329759633
// MI455X (gfx1250) — matrix operand order repaired and checked by reading, not compiled, not run on hardware
//
#include <hip/hip_runtime.h>
#include <hip/hip_fp16.h>
#include <stdint.h>

#define EMS 2048
#define NC  128
#define NS  32768

#if __has_builtin(__builtin_amdgcn_global_load_async_to_lds_b128)
#define ASYNC_LDS 1
#else
#define ASYNC_LDS 0
#endif

typedef __attribute__((ext_vector_type(16))) _Float16 v16h;
typedef __attribute__((ext_vector_type(8)))  _Float16 v8h;
typedef __attribute__((ext_vector_type(8)))  float    v8f;
typedef int v4i __attribute__((__vector_size__(4 * sizeof(int))));

#define AS_GLOBAL __attribute__((address_space(1)))
#define AS_LOCAL  __attribute__((address_space(3)))

__device__ __forceinline__ uint32_t pack2h(float a, float b) {
    __half2 h = __floats2half2_rn(a, b);
    return *reinterpret_cast<uint32_t*>(&h);
}
__device__ __forceinline__ v8f vzero8() {
    v8f v;
#pragma unroll
    for (int i = 0; i < 8; ++i) v[i] = 0.f;
    return v;
}
__device__ __forceinline__ v16h frag16(const uint32_t* p) {
    // 8 consecutive dwords -> v16h (two b128 loads, no register shuffling)
    v8h lo = *reinterpret_cast<const v8h*>(p);
    v8h hi = *reinterpret_cast<const v8h*>(p + 4);
    return __builtin_shufflevector(lo, hi, 0, 1, 2, 3, 4, 5, 6, 7,
                                   8, 9, 10, 11, 12, 13, 14, 15);
}
__device__ __forceinline__ void wait_async0() {
#if ASYNC_LDS
#if __has_builtin(__builtin_amdgcn_s_wait_asynccnt)
    __builtin_amdgcn_s_wait_asynccnt(0);
#else
    asm volatile("s_wait_asynccnt 0" ::: "memory");
#endif
#endif
}

// ---------------------------------------------------------------------------
// Pack weights to f16, K-pair packed, FRAGMENT-MAJOR:
// W1p: [kc 0..63][n 0..383][kp 0..15]  (n = tap*128 + c; K = 2*(kc*16+kp))
//      -> each column's 16 chunk-dwords are contiguous (B frag = 2 x b128).
// W2p: [tap 0..2][kc 0..3][n 0..127][q 0..15]
// ---------------------------------------------------------------------------
__global__ __launch_bounds__(256) void pack_w_kernel(const float* __restrict__ W1,
                                                     const float* __restrict__ W2,
                                                     uint32_t* __restrict__ W1p,
                                                     uint32_t* __restrict__ W2p) {
    int idx = blockIdx.x * 256 + threadIdx.x;
    const int NW1 = 64 * 384 * 16;     // 393216
    const int NW2 = 3 * 4 * 128 * 16;  // 24576
    if (idx < NW1) {
        int kp = idx & 15;
        int n  = (idx >> 4) % 384;
        int kc = idx / 6144;
        int tap = n >> 7, c = n & 127;
        int e = 2 * (kc * 16 + kp);
        const float* s = W1 + (size_t)tap * EMS * NC + (size_t)e * NC + c;
        W1p[idx] = pack2h(s[0], s[NC]);
    } else if (idx < NW1 + NW2) {
        int j2 = idx - NW1;
        int q   = j2 & 15;
        int n   = (j2 >> 4) & 127;
        int kc  = (j2 >> 11) & 3;
        int tap = j2 >> 13;
        int e = 2 * (kc * 16 + q);
        const float* s = W2 + (size_t)tap * NC * NC + (size_t)e * NC + n;
        W2p[j2] = pack2h(s[0], s[NC]);
    }
}

// ---------------------------------------------------------------------------
// Fused Stage 1: S[p] = sum_k x[p-k] @ W1[k] (A-row shift per tap, 2-row halo),
// H[p] = tanh(S + cnt1(p)*b1) -> f16. Double-buffered, async B staging.
// ---------------------------------------------------------------------------
#define ASTR 20    // A LDS row stride (dwords): 16B aligned, full bank permutation
#define BSTR 20    // B LDS per-column stride (dwords): 16 kp dwords + 4 pad
#define AROWS 130
#define AITEMS (AROWS * 16)  // 2080
#define AIT 9

__global__ __launch_bounds__(256) void s1_fused_kernel(const float* __restrict__ x,
                                                       const uint32_t* __restrict__ W1p,
                                                       const float* __restrict__ b1,
                                                       uint32_t* __restrict__ Hp) {
    __shared__ uint32_t Alds[2][AROWS * ASTR]; // 2 x 10.2 KB
    __shared__ uint32_t Blds[2][384 * BSTR];   // 2 x 30.0 KB
    const int tid  = threadIdx.x;
    const int wave = tid >> 5;
    const int lane = tid & 31;
    const int lh   = lane & 15;
    const int hi   = lane >> 4;
    const int m0   = blockIdx.x * 128;
    const int wm   = wave & 1;   // M group (4 tiles each)
    const int wn   = wave >> 1;  // N group (2 tiles each)

    v8f acc[4][2];
#pragma unroll
    for (int a = 0; a < 4; ++a)
#pragma unroll
        for (int b = 0; b < 2; ++b) acc[a][b] = vzero8();

    float2 fa[AIT];
#if !ASYNC_LDS
    uint4 fb[6];
#endif

    auto loadA = [&](int kc) {
#pragma unroll
        for (int i = 0; i < AIT; ++i) {
            int idx = tid + i * 256;
            int kp = idx & 15, lr = idx >> 4;
            int gr = m0 - 2 + lr;
            // i==0: halo rows may be <0 (block 0). i==8: tail beyond 2080
            // (and gr beyond NS on the last block). i=1..7: always valid.
            bool valid = (i == 0) ? (gr >= 0) : (i < 8 ? true : (idx < AITEMS));
            int grc = valid ? gr : 0;  // clamp to a safe row, zero after load
            float2 v = *reinterpret_cast<const float2*>(
                x + (size_t)grc * EMS + kc * 32 + kp * 2);
            if (!valid) { v.x = 0.f; v.y = 0.f; }
            fa[i] = v;
        }
    };
    auto storeA = [&](int buf) {
#pragma unroll
        for (int i = 0; i < AIT; ++i) {
            int idx = tid + i * 256;
            if (i < 8 || idx < AITEMS) {
                int kp = idx & 15, lr = idx >> 4;
                Alds[buf][lr * ASTR + kp] = pack2h(fa[i].x, fa[i].y);
            }
        }
    };
    auto issueB = [&](int kc, int buf) {
        const uint32_t* bs = W1p + (size_t)kc * 6144;
#pragma unroll
        for (int i = 0; i < 6; ++i) {
            int g = tid + i * 256;        // 16B group: col n = g>>2, quarter q = g&3
            int n = g >> 2, q = g & 3;
#if ASYNC_LDS
            __builtin_amdgcn_global_load_async_to_lds_b128(
                (AS_GLOBAL v4i*)(bs + g * 4),
                (AS_LOCAL  v4i*)&Blds[buf][n * BSTR + q * 4],
                0, 0);
#else
            fb[i] = *reinterpret_cast<const uint4*>(bs + g * 4);
#endif
        }
    };
#if !ASYNC_LDS
    auto storeB = [&](int buf) {
#pragma unroll
        for (int i = 0; i < 6; ++i) {
            int g = tid + i * 256;
            int n = g >> 2, q = g & 3;
            *reinterpret_cast<uint4*>(&Blds[buf][n * BSTR + q * 4]) = fb[i];
        }
    };
#endif
    auto compute = [&](int buf) {
#pragma unroll
        for (int k = 0; k < 3; ++k) {  // tap -> A-row shift (2-k)
            v16h af[4];
#pragma unroll
            for (int mtl = 0; mtl < 4; ++mtl) {
                int lr = (wm * 4 + mtl) * 16 + lh + 2 - k;
                const uint32_t* ab = &Alds[buf][lr * ASTR];
                v8h lo  = *reinterpret_cast<const v8h*>(ab + hi * 8);
                v8h hi8 = *reinterpret_cast<const v8h*>(ab + 4 + hi * 8);
                af[mtl] = __builtin_shufflevector(lo, hi8, 0, 1, 2, 3, 4, 5, 6, 7,
                                                  8, 9, 10, 11, 12, 13, 14, 15);
            }
#pragma unroll
            for (int ntl = 0; ntl < 2; ++ntl) {
                int col = k * 128 + (wn * 2 + ntl) * 16 + lh;
                v16h bv = frag16(&Blds[buf][col * BSTR + hi * 8]);
#pragma unroll
                for (int mtl = 0; mtl < 4; ++mtl)
                    acc[mtl][ntl] = __builtin_amdgcn_wmma_f32_16x16x32_f16(
                        false, af[mtl], false, bv, (short)0, acc[mtl][ntl],
                        false, false);
            }
        }
    };

    // ---- pipeline prologue
    loadA(0);
    issueB(0, 0);
    storeA(0);
#if !ASYNC_LDS
    storeB(0);
#endif
    wait_async0();
    __syncthreads();

#pragma unroll 2
    for (int kc = 0; kc < EMS / 32; ++kc) {
        int cur = kc & 1;
        if (kc < EMS / 32 - 1) { loadA(kc + 1); issueB(kc + 1, cur ^ 1); }
        compute(cur);
        if (kc < EMS / 32 - 1) {
            storeA(cur ^ 1);
#if !ASYNC_LDS
            storeB(cur ^ 1);
#endif
        }
        wait_async0();
        __syncthreads();
    }

    // ---- epilogue: H = tanh(S + cnt1*b1) -> Hp (f16 row-major, dword pairs)
#pragma unroll
    for (int mtl = 0; mtl < 4; ++mtl) {
#pragma unroll
        for (int ntl = 0; ntl < 2; ++ntl) {
            int col = (wn * 2 + ntl) * 16 + lh;
            float bv = b1[col];
#pragma unroll
            for (int r = 0; r < 8; ++r) {
                int row = m0 + (wm * 4 + mtl) * 16 + r + hi * 8;
                float cnt = (row >= 2) ? 3.f : (row >= 1 ? 2.f : 1.f);
                float h  = tanhf(acc[mtl][ntl][r] + cnt * bv);
                float hx = __shfl_xor(h, 1, 32);   // neighbor col (all lanes exec)
                if (!(lh & 1))
                    Hp[(size_t)row * 64 + (col >> 1)] = pack2h(h, hx);
            }
        }
    }
}

// ---------------------------------------------------------------------------
// Stage 2 + log_softmax: Z[i] = sum_j H[i-2+j] @ W2[j] + 3*b2
// Each wave: one 16-row M-tile x all 8 N-tiles -> softmax is wave-local.
// B fragments come fragment-major from L2 (2 x global b128 per frag).
// ---------------------------------------------------------------------------
#define HSTR 68  // 272B, 16B aligned; conflict-free b128 row gathers

__global__ __launch_bounds__(256) void s2_kernel(const uint32_t* __restrict__ Hp,
                                                 const uint32_t* __restrict__ W2p,
                                                 const float* __restrict__ b2,
                                                 float* __restrict__ out) {
    __shared__ uint32_t Hlds[130 * HSTR];
    const int tid  = threadIdx.x;
    const int wave = tid >> 5;
    const int lane = tid & 31;
    const int lh   = lane & 15;
    const int hi   = lane >> 4;
    const int m0   = blockIdx.x * 128;

    for (int i = 0; i < 33; ++i) {
        int idx = tid + i * 256;
        if (idx < 130 * 64) {
            int cp = idx & 63;
            int lr = idx >> 6;
            int gr = m0 - 2 + lr;
            Hlds[lr * HSTR + cp] = (gr >= 0) ? Hp[(size_t)gr * 64 + cp] : 0u;
        }
    }
    __syncthreads();

    v8f acc[8];
#pragma unroll
    for (int i = 0; i < 8; ++i) acc[i] = vzero8();

#pragma unroll
    for (int j = 0; j < 3; ++j) {
#pragma unroll
        for (int kc = 0; kc < 4; ++kc) {
            const uint32_t* ab = &Hlds[(wave * 16 + lh + j) * HSTR + kc * 16];
            v8h lo  = *reinterpret_cast<const v8h*>(ab + hi * 8);
            v8h hi8 = *reinterpret_cast<const v8h*>(ab + 4 + hi * 8);
            v16h af = __builtin_shufflevector(lo, hi8, 0, 1, 2, 3, 4, 5, 6, 7,
                                              8, 9, 10, 11, 12, 13, 14, 15);
            const uint32_t* wb = W2p + (size_t)j * 8192 + (size_t)kc * 2048;
#pragma unroll
            for (int nt = 0; nt < 8; ++nt) {
                v16h bv = frag16(wb + (size_t)(nt * 16 + lh) * 16 + hi * 8);
                acc[nt] = __builtin_amdgcn_wmma_f32_16x16x32_f16(
                    false, af, false, bv, (short)0, acc[nt], false, false);
            }
        }
    }

    float b2v[8];
#pragma unroll
    for (int nt = 0; nt < 8; ++nt) b2v[nt] = 3.f * b2[nt * 16 + lh];

#pragma unroll
    for (int r = 0; r < 8; ++r) {
        float zv[8];
        float m = -3.4e38f;
#pragma unroll
        for (int nt = 0; nt < 8; ++nt) { zv[nt] = acc[nt][r] + b2v[nt]; m = fmaxf(m, zv[nt]); }
#pragma unroll
        for (int s = 1; s < 16; s <<= 1) m = fmaxf(m, __shfl_xor(m, s, 32));
        float sum = 0.f;
#pragma unroll
        for (int nt = 0; nt < 8; ++nt) sum += expf(zv[nt] - m);
#pragma unroll
        for (int s = 1; s < 16; s <<= 1) sum += __shfl_xor(sum, s, 32);
        float lse = m + logf(sum);
        int row = m0 + wave * 16 + r + hi * 8;
#pragma unroll
        for (int nt = 0; nt < 8; ++nt) out[(size_t)row * NC + nt * 16 + lh] = zv[nt] - lse;
    }
}

// ---------------------------------------------------------------------------
// Fallback rows 0,1: Z0 = tanh(x[i]@W1[0]+b1) @ W2[0] + b2, log_softmax.
// ---------------------------------------------------------------------------
__global__ __launch_bounds__(128) void fallback_kernel(const float* __restrict__ x,
                                                       const float* __restrict__ W1,
                                                       const float* __restrict__ b1,
                                                       const float* __restrict__ W2,
                                                       const float* __restrict__ b2,
                                                       float* __restrict__ out) {
    __shared__ float h0[NC];
    __shared__ float zb[NC];
    int c = threadIdx.x;
    for (int i = 0; i < 2; ++i) {
        float s[8];
#pragma unroll
        for (int u = 0; u < 8; ++u) s[u] = 0.f;
        for (int e = 0; e < EMS; e += 8)
#pragma unroll
            for (int u = 0; u < 8; ++u)
                s[u] += x[(size_t)i * EMS + e + u] * W1[(size_t)(e + u) * NC + c];
        float sv = ((s[0] + s[1]) + (s[2] + s[3])) + ((s[4] + s[5]) + (s[6] + s[7]));
        h0[c] = tanhf(sv + b1[c]);
        __syncthreads();
        float z = b2[c];
        for (int q = 0; q < NC; ++q) z += h0[q] * W2[(size_t)q * NC + c];
        zb[c] = z;
        __syncthreads();
        float m = -3.4e38f;
        for (int q = 0; q < NC; ++q) m = fmaxf(m, zb[q]);
        float ssum = 0.f;
        for (int q = 0; q < NC; ++q) ssum += expf(zb[q] - m);
        out[(size_t)i * NC + c] = z - m - logf(ssum);
        __syncthreads();
    }
}

// ---------------------------------------------------------------------------
extern "C" void kernel_launch(void* const* d_in, const int* in_sizes, int n_in,
                              void* d_out, int out_size, void* d_ws, size_t ws_size,
                              hipStream_t stream) {
    (void)in_sizes; (void)n_in; (void)out_size; (void)ws_size;
    const float* x  = (const float*)d_in[0];
    const float* W1 = (const float*)d_in[1];
    const float* b1 = (const float*)d_in[2];
    const float* W2 = (const float*)d_in[3];
    const float* b2 = (const float*)d_in[4];
    float* out = (float*)d_out;

    char* ws = (char*)d_ws;
    uint32_t* Hp  = (uint32_t*)ws;                                  // 8 MB
    uint32_t* W1p = (uint32_t*)(ws + (size_t)NS * (NC / 2) * 4);    // 1.5 MB
    uint32_t* W2p = W1p + (size_t)64 * 384 * 16;                    // 96 KB

    pack_w_kernel<<<1632, 256, 0, stream>>>(W1, W2, W1p, W2p);
    s1_fused_kernel<<<NS / 128, 256, 0, stream>>>(x, W1p, b1, Hp);
    s2_kernel<<<NS / 128, 256, 0, stream>>>(Hp, W2p, b2, out);
    fallback_kernel<<<1, 128, 0, stream>>>(x, W1, b1, W2, b2, out);
}
